// Attention_73254962200646
// MI455X (gfx1250) — matrix operand order repaired and checked by reading, not compiled, not run on hardware
//
#include <hip/hip_runtime.h>
#include <hip/hip_bf16.h>

typedef _Float16 half_t;
typedef __attribute__((ext_vector_type(16))) _Float16 v16h;
typedef __attribute__((ext_vector_type(8)))  _Float16 v8h;
typedef __attribute__((ext_vector_type(8)))  float    v8f;

#define DIMC   512
#define BATCH  8
#define HWN    1024          // 32*32
#define NTOT   8192          // BATCH*HWN
#define KC     4608          // DIMC*9
#define SMOOTHF 1e-4f
#define BN_EPSF 1e-5f

__device__ __forceinline__ v8f wmma16(v16h a, v16h b, v8f c) {
  return __builtin_amdgcn_wmma_f32_16x16x32_f16(false, a, false, b, (short)0, c, false, false);
}

__device__ __forceinline__ v8f zero8() {
  v8f z;
#pragma unroll
  for (int e = 0; e < 8; ++e) z[e] = 0.0f;
  return z;
}

// A-fragment: lane holds row `rowptr`, 16 contiguous f16 at k=2*klo..2*klo+15 (contiguous k order, matches B)
__device__ __forceinline__ v16h pack_a(const half_t* rowptr, int klo) {
  v8h lo = *(const v8h*)(rowptr + klo * 2);
  v8h hi = *(const v8h*)(rowptr + klo * 2 + 8);
  v16h r;
#pragma unroll
  for (int e = 0; e < 8; ++e) { r[e] = lo[e]; r[e + 8] = hi[e]; }
  return r;
}

// Async global->LDS copy of 16 bytes (per-lane). LDS byte address = low 32 bits of the
// generic pointer to the __shared__ destination (LDS aperture truncation rule, ISA 10.2).
__device__ __forceinline__ void async_copy16(void* lds_ptr, const void* gptr) {
  unsigned off = (unsigned)(unsigned long long)lds_ptr;
  asm volatile("global_load_async_to_lds_b128 %0, %1, off"
               :: "v"(off), "v"(gptr) : "memory");
}

__device__ __forceinline__ void wait_async0() {
  asm volatile("s_wait_asynccnt 0x0" ::: "memory");
}

// ---------------------------------------------------------------- prep: f32 -> f16 casts + zero BN accumulators
__global__ __launch_bounds__(256) void prep_kernel(
    const float* __restrict__ x, const float* __restrict__ wqkv, const float* __restrict__ wout,
    half_t* __restrict__ xh, half_t* __restrict__ wqkvh, half_t* __restrict__ wouth,
    float* __restrict__ chsum, float* __restrict__ chsq) {
  size_t idx = (size_t)blockIdx.x * 256 + threadIdx.x;
  if (idx < (size_t)BATCH * DIMC * HWN) xh[idx] = (half_t)x[idx];
  if (idx < (size_t)3 * DIMC * KC)      wqkvh[idx] = (half_t)wqkv[idx];
  if (idx < (size_t)DIMC * DIMC)        wouth[idx] = (half_t)wout[idx];
  if (idx < DIMC) { chsum[idx] = 0.0f; chsq[idx] = 0.0f; }
}

// ---------------------------------------------------------------- 3x3 conv as implicit GEMM (WMMA)
// C[oc, n] = sum_k W[oc,k] * im2col(x)[k, n];  M=1536, N=8192, K=4608
__global__ __launch_bounds__(256) void conv3x3_kernel(
    const half_t* __restrict__ xh, const half_t* __restrict__ wh, half_t* __restrict__ qkvh) {
  __shared__ __align__(32) half_t As[128 * 32];   // [m][k]
  __shared__ __align__(32) half_t Bs[128 * 32];   // [n][k] (transposed for contiguous B-frag reads)
  const int tid = threadIdx.x;
  const int lane = tid & 31, wid = tid >> 5;
  const int waveM = wid >> 1, waveN = wid & 1;
  const int khalf = lane >> 4, l15 = lane & 15;
  const int M0 = blockIdx.y * 128, N0 = blockIdx.x * 128;

  const int nloc = tid & 127;
  const int kblk = (tid >> 7) * 16;
  const int nglb = N0 + nloc;
  const int bb = nglb >> 10;
  const int hw = nglb & 1023;
  const int h0 = hw >> 5, w0 = hw & 31;

  const size_t arow = (size_t)(M0 + (tid >> 1)) * KC + (size_t)((tid & 1) * 16);
  half_t* asdst = &As[(tid >> 1) * 32 + (tid & 1) * 16];

  v8f acc[2][4];
#pragma unroll
  for (int mi = 0; mi < 2; ++mi)
#pragma unroll
    for (int ni = 0; ni < 4; ++ni) acc[mi][ni] = zero8();

  for (int k0 = 0; k0 < KC; k0 += 32) {
    { // A tile: async global->LDS, two 16B chunks per thread
      const half_t* src = wh + arow + k0;
      async_copy16(asdst, src);
      async_copy16(asdst + 8, src + 8);
      __builtin_prefetch(src + 32, 0, 0);   // next k-step's weights
    }
    { // B tile: im2col gather, 16 elems per thread, contiguous LDS write
      half_t tmp[16];
#pragma unroll
      for (int kk = 0; kk < 16; ++kk) {
        int k = k0 + kblk + kk;
        int ic = k / 9;
        int r = k - ic * 9;
        int kh = r / 3;
        int kw = r - kh * 3;
        int hh = h0 + kh - 1;
        int ww = w0 + kw - 1;
        half_t v = (half_t)0.0f;
        if ((unsigned)hh < 32u && (unsigned)ww < 32u)
          v = xh[(((size_t)bb * DIMC + ic) << 10) + (hh << 5) + ww];
        tmp[kk] = v;
      }
      v8h t0, t1;
#pragma unroll
      for (int e = 0; e < 8; ++e) { t0[e] = tmp[e]; t1[e] = tmp[e + 8]; }
      v8h* dst = (v8h*)&Bs[nloc * 32 + kblk];
      dst[0] = t0; dst[1] = t1;
    }
    wait_async0();
    __syncthreads();
#pragma unroll
    for (int mi = 0; mi < 2; ++mi) {
      v16h a = pack_a(&As[(waveM * 32 + mi * 16 + l15) * 32], khalf * 8);
#pragma unroll
      for (int ni = 0; ni < 4; ++ni) {
        v16h b = *(const v16h*)&Bs[(waveN * 64 + ni * 16 + l15) * 32 + khalf * 16];
        acc[mi][ni] = wmma16(a, b, acc[mi][ni]);
      }
    }
    __syncthreads();
  }

#pragma unroll
  for (int mi = 0; mi < 2; ++mi)
#pragma unroll
    for (int ni = 0; ni < 4; ++ni)
#pragma unroll
      for (int v = 0; v < 8; ++v) {
        int oc = M0 + waveM * 32 + mi * 16 + khalf * 8 + v;
        int col = N0 + waveN * 64 + ni * 16 + l15;
        qkvh[(size_t)oc * NTOT + col] = (half_t)acc[mi][ni][v];
      }
}

// ---------------------------------------------------------------- repack q,v to row-major [head*1024+n][64], norms
__global__ __launch_bounds__(256) void repack_kernel(
    const half_t* __restrict__ qkvh, half_t* __restrict__ Qh, half_t* __restrict__ Vh,
    float* __restrict__ qn, float* __restrict__ kn) {
  int idx = blockIdx.x * 256 + threadIdx.x;   // 65536 = b*g*n
  int n = idx & 1023;
  int g = (idx >> 10) & 7;
  int b = idx >> 13;
  size_t col = (size_t)b * 1024 + n;
  size_t rowoff = (size_t)idx * 64;
  float sq = 0.0f, sk = 0.0f;
  for (int d = 0; d < 64; ++d) {
    size_t ch = (size_t)(g * 64 + d);
    half_t qv = qkvh[ch * NTOT + col];
    half_t kv = qkvh[(ch + 512) * NTOT + col];
    half_t vv = qkvh[(ch + 1024) * NTOT + col];
    Qh[rowoff + d] = qv;
    Vh[rowoff + d] = vv;
    float qf = (float)qv, kf = (float)kv;
    sq += qf * qf;
    sk += kf * kf;
  }
  qn[idx] = sqrtf(sq + SMOOTHF);
  kn[idx] = sqrtf(sk + SMOOTHF);
}

// ---------------------------------------------------------------- attention: S = QK^T, normalize, O = S V  (WMMA)
__global__ __launch_bounds__(256) void attn_kernel(
    const half_t* __restrict__ Qh, const half_t* __restrict__ qkvh, const half_t* __restrict__ Vh,
    const float* __restrict__ qn, const float* __restrict__ kn, half_t* __restrict__ Oh) {
  __shared__ __align__(32) half_t Qs[128 * 64];   // [i][d]
  __shared__ __align__(32) half_t Ks[32 * 64];    // [j][d]
  __shared__ __align__(32) half_t Vs[64 * 32];    // [d][j]
  __shared__ __align__(32) half_t Ss[8 * 16 * 32]; // per-wave S tile [i16][j32]
  __shared__ float qnS[128];
  __shared__ float knS[1024];

  const int blk = blockIdx.x;
  const int head = blk >> 3, itile = blk & 7;
  const int b = head >> 3, g = head & 7;
  const int tid = threadIdx.x, lane = tid & 31, wid = tid >> 5;
  const int khalf = lane >> 4, l15 = lane & 15;
  const size_t headQ = (size_t)head * 1024 * 64;
  const int i0 = itile * 128;

  { // Q tile for this i-block: 128x64, async global->LDS (4x16B per thread)
    int row = tid >> 1, hs = (tid & 1) * 32;
    const half_t* src = Qh + headQ + (size_t)(i0 + row) * 64 + hs;
    half_t* dst = &Qs[row * 64 + hs];
    async_copy16(dst,      src);
    async_copy16(dst + 8,  src + 8);
    async_copy16(dst + 16, src + 16);
    async_copy16(dst + 24, src + 24);
  }
  if (tid < 128) qnS[tid] = qn[(size_t)head * 1024 + i0 + tid];
  for (int u = tid; u < 1024; u += 256) knS[u] = kn[(size_t)head * 1024 + u];
  wait_async0();
  __syncthreads();

  // Per-wave Q fragments (rows wid*16..+16, full d=64) and qn values — live across whole j loop
  v16h qa[2];
  const half_t* qrow = &Qs[(wid * 16 + l15) * 64];
  qa[0] = pack_a(qrow, khalf * 8);
  qa[1] = pack_a(qrow + 32, khalf * 8);
  float qn_r[8];
#pragma unroll
  for (int v = 0; v < 8; ++v) qn_r[v] = qnS[wid * 16 + khalf * 8 + v];

  v8f oacc[4];
#pragma unroll
  for (int dn = 0; dn < 4; ++dn) oacc[dn] = zero8();

  half_t* swave = &Ss[wid * 512];

  for (int jb = 0; jb < 1024; jb += 32) {
    __syncthreads();
    { // K tile [j][d] from qkvh channel-major K section (coalesced along j)
      int d = tid >> 2, jc = (tid & 3) * 8;
      const half_t* src = qkvh + ((size_t)(512 + g * 64 + d)) * NTOT + (size_t)b * 1024 + jb + jc;
      v8h kv = *(const v8h*)src;
#pragma unroll
      for (int e = 0; e < 8; ++e) Ks[(jc + e) * 64 + d] = kv[e];
    }
    { // V tile transposed to [d][j] (coalesced along d)
      int j = tid >> 3, dc = (tid & 7) * 8;
      const v8h vv = *(const v8h*)(Vh + headQ + (size_t)(jb + j) * 64 + dc);
#pragma unroll
      for (int e = 0; e < 8; ++e) Vs[(dc + e) * 32 + j] = vv[e];
    }
    __syncthreads();

    // S block 16i x 32j, accumulate over d=64, normalize, stash to LDS as f16
#pragma unroll
    for (int jn = 0; jn < 2; ++jn) {
      v8f s = zero8();
#pragma unroll
      for (int kd = 0; kd < 2; ++kd) {
        v16h bk = *(const v16h*)&Ks[(jn * 16 + l15) * 64 + kd * 32 + khalf * 16];
        s = wmma16(qa[kd], bk, s);
      }
      float knj = knS[jb + jn * 16 + l15];
#pragma unroll
      for (int v = 0; v < 8; ++v) {
        float sn = s[v] / (qn_r[v] * knj + SMOOTHF);
        swave[(khalf * 8 + v) * 32 + jn * 16 + l15] = (half_t)sn;
      }
    }

    // O += S_blk * V_blk : A from Ss (A-layout), B from Vs
    v16h sa = pack_a(&swave[l15 * 32], khalf * 8);
#pragma unroll
    for (int dn = 0; dn < 4; ++dn) {
      v16h bv = *(const v16h*)&Vs[(dn * 16 + l15) * 32 + khalf * 16];
      oacc[dn] = wmma16(sa, bv, oacc[dn]);
    }
  }

  // Store O in channel-major layout [g*64+d][b*1024+i] for the 1x1 conv GEMM
#pragma unroll
  for (int dn = 0; dn < 4; ++dn)
#pragma unroll
    for (int v = 0; v < 8; ++v) {
      int d = dn * 16 + l15;
      int i = i0 + wid * 16 + khalf * 8 + v;
      Oh[((size_t)(g * 64 + d)) * NTOT + (size_t)b * 1024 + i] = (half_t)oacc[dn][v];
    }
}

// ---------------------------------------------------------------- 1x1 conv GEMM (WMMA) + fused BN partial sums
__global__ __launch_bounds__(256) void gemm1x1_kernel(
    const half_t* __restrict__ wh, const half_t* __restrict__ Oh,
    float* __restrict__ y, float* __restrict__ chsum, float* __restrict__ chsq) {
  __shared__ __align__(32) half_t As[128 * 32];
  __shared__ __align__(32) half_t Bs[128 * 32];   // [n][k]
  const int tid = threadIdx.x, lane = tid & 31, wid = tid >> 5;
  const int waveM = wid >> 1, waveN = wid & 1;
  const int khalf = lane >> 4, l15 = lane & 15;
  const int M0 = blockIdx.y * 128, N0 = blockIdx.x * 128;

  const size_t arow = (size_t)(M0 + (tid >> 1)) * DIMC + (size_t)((tid & 1) * 16);
  half_t* asdst = &As[(tid >> 1) * 32 + (tid & 1) * 16];
  const int kl = tid >> 3;          // 0..31
  const int nb = (tid & 7) * 16;    // 0..112

  v8f acc[2][4];
#pragma unroll
  for (int mi = 0; mi < 2; ++mi)
#pragma unroll
    for (int ni = 0; ni < 4; ++ni) acc[mi][ni] = zero8();

  for (int k0 = 0; k0 < DIMC; k0 += 32) {
    { // A tile: async global->LDS
      const half_t* src = wh + arow + k0;
      async_copy16(asdst, src);
      async_copy16(asdst + 8, src + 8);
    }
    { // B tile: transposed scatter into [n][k]
      const half_t* src = Oh + (size_t)(k0 + kl) * NTOT + N0 + nb;
      v8h b0 = ((const v8h*)src)[0], b1 = ((const v8h*)src)[1];
#pragma unroll
      for (int e = 0; e < 8; ++e) {
        Bs[(nb + e) * 32 + kl] = b0[e];
        Bs[(nb + 8 + e) * 32 + kl] = b1[e];
      }
    }
    wait_async0();
    __syncthreads();
#pragma unroll
    for (int mi = 0; mi < 2; ++mi) {
      v16h a = pack_a(&As[(waveM * 32 + mi * 16 + l15) * 32], khalf * 8);
#pragma unroll
      for (int ni = 0; ni < 4; ++ni) {
        v16h b = *(const v16h*)&Bs[(waveN * 64 + ni * 16 + l15) * 32 + khalf * 16];
        acc[mi][ni] = wmma16(a, b, acc[mi][ni]);
      }
    }
    __syncthreads();
  }

#pragma unroll
  for (int mi = 0; mi < 2; ++mi)
#pragma unroll
    for (int v = 0; v < 8; ++v) {
      int oc = M0 + waveM * 32 + mi * 16 + khalf * 8 + v;
      float s = 0.0f, sq = 0.0f;
#pragma unroll
      for (int ni = 0; ni < 4; ++ni) {
        float val = acc[mi][ni][v];
        int col = N0 + waveN * 64 + ni * 16 + l15;
        y[(size_t)oc * NTOT + col] = val;
        s += val; sq += val * val;
      }
#pragma unroll
      for (int m = 1; m < 16; m <<= 1) {
        s  += __shfl_xor(s, m, 32);
        sq += __shfl_xor(sq, m, 32);
      }
      if (l15 == 0) { atomicAdd(&chsum[oc], s); atomicAdd(&chsq[oc], sq); }
    }
}

// ---------------------------------------------------------------- BN (batch stats) + affine + ReLU, NCHW output
__global__ __launch_bounds__(256) void bn_relu_kernel(
    const float* __restrict__ y, const float* __restrict__ chsum, const float* __restrict__ chsq,
    const float* __restrict__ gamma, const float* __restrict__ beta, float* __restrict__ out) {
  size_t idx = (size_t)blockIdx.x * 256 + threadIdx.x;   // b*c*hw flat, 4194304
  int hw = (int)(idx & 1023);
  int c = (int)((idx >> 10) & 511);
  int b = (int)(idx >> 19);
  const float inv_n = 1.0f / 8192.0f;
  float mean = chsum[c] * inv_n;
  float var = chsq[c] * inv_n - mean * mean;
  float sc = gamma[c] * rsqrtf(var + BN_EPSF);
  float val = (y[(size_t)c * NTOT + (size_t)b * 1024 + hw] - mean) * sc + beta[c];
  out[idx] = fmaxf(val, 0.0f);
}

// ---------------------------------------------------------------- host launcher
extern "C" void kernel_launch(void* const* d_in, const int* in_sizes, int n_in,
                              void* d_out, int out_size, void* d_ws, size_t ws_size,
                              hipStream_t stream) {
  (void)in_sizes; (void)n_in; (void)out_size; (void)ws_size;
  const float* x     = (const float*)d_in[0];
  const float* wqkv  = (const float*)d_in[1];
  const float* wout  = (const float*)d_in[2];
  const float* gamma = (const float*)d_in[3];
  const float* beta  = (const float*)d_in[4];
  float* out = (float*)d_out;

  char* p = (char*)d_ws;
  auto alloc = [&](size_t bytes) { char* r = p; p += (bytes + 255) & ~(size_t)255; return r; };
  half_t* xh    = (half_t*)alloc((size_t)BATCH * DIMC * HWN * 2);   // x in f16
  half_t* wqkvh = (half_t*)alloc((size_t)3 * DIMC * KC * 2);        // W_qkv f16 [1536][4608]
  half_t* wouth = (half_t*)alloc((size_t)DIMC * DIMC * 2);          // W_out f16 [512][512]
  half_t* qkvh  = (half_t*)alloc((size_t)3 * DIMC * NTOT * 2);      // conv out f16 [1536][8192]
  half_t* Qh    = (half_t*)alloc((size_t)64 * 1024 * 64 * 2);       // Q row-major per head
  half_t* Vh    = (half_t*)alloc((size_t)64 * 1024 * 64 * 2);       // V row-major per head
  float*  qn    = (float*)alloc((size_t)64 * 1024 * 4);
  float*  kn    = (float*)alloc((size_t)64 * 1024 * 4);
  half_t* Oh    = (half_t*)alloc((size_t)DIMC * NTOT * 2);          // attn out channel-major
  float*  y     = (float*)alloc((size_t)DIMC * NTOT * 4);           // 1x1 conv out f32
  float*  chsum = (float*)alloc(512 * 4);
  float*  chsq  = (float*)alloc(512 * 4);

  prep_kernel<<<27648, 256, 0, stream>>>(x, wqkv, wout, xh, wqkvh, wouth, chsum, chsq);
  conv3x3_kernel<<<dim3(64, 12), 256, 0, stream>>>(xh, wqkvh, qkvh);
  repack_kernel<<<256, 256, 0, stream>>>(qkvh, Qh, Vh, qn, kn);
  attn_kernel<<<512, 256, 0, stream>>>(Qh, qkvh, Vh, qn, kn, Oh);
  gemm1x1_kernel<<<dim3(64, 4), 256, 0, stream>>>(wouth, Oh, y, chsum, chsq);
  bn_relu_kernel<<<16384, 256, 0, stream>>>(y, chsum, chsq, gamma, beta, out);
}
